// MOLELinear_45586782880338
// MI455X (gfx1250) — compile-verified, matrix operand order checked by reading, not run on hardware
//
#include <hip/hip_runtime.h>
#include <hip/hip_bf16.h>

// ---------------- problem constants (static from reference) ----------------
#define IN_F   512
#define OUT_F  512
#define NEXP   8
#define NGRP   32
#define TPG    2048
#define NTOK   (NGRP * TPG)      // 65536

// ---------------- GEMM tiling ----------------
#define BM  128
#define BN  128
#define BK  64
#define LDT 80                   // LDS row stride in bf16 (160B: 32B-aligned frags, conflict-padded)

typedef __attribute__((ext_vector_type(16))) __bf16 bf16x16;
typedef __attribute__((ext_vector_type(8)))  __bf16 bf16x8;
typedef __attribute__((ext_vector_type(8)))  float  f32x8;
typedef __attribute__((ext_vector_type(4)))  float  f32x4;

#define ASYNC_WAIT0() asm volatile("s_wait_asynccnt 0x0" ::: "memory")

// per-lane async DMA of 16 bytes: global -> LDS, tracked by ASYNCcnt
__device__ __forceinline__ void async_cp16(unsigned lds_byte_addr, const void* gptr) {
  asm volatile("global_load_async_to_lds_b128 %0, %1, off"
               :: "v"(lds_byte_addr), "v"(gptr)
               : "memory");
}

// ---------------------------------------------------------------------------
// Kernel 1: convert x (f32) -> bf16, 8 elements/thread
// ---------------------------------------------------------------------------
__global__ __launch_bounds__(256) void mole_xcvt(const float* __restrict__ x,
                                                 unsigned short* __restrict__ xb_) {
  __bf16* xb = reinterpret_cast<__bf16*>(xb_);
  size_t idx = ((size_t)blockIdx.x * 256 + threadIdx.x) * 8;
  f32x4 a = *(const f32x4*)(x + idx);
  f32x4 b = *(const f32x4*)(x + idx + 4);
  bf16x8 o;
#pragma unroll
  for (int i = 0; i < 4; ++i) { o[i] = (__bf16)a[i]; o[i + 4] = (__bf16)b[i]; }
  *(bf16x8*)(xb + idx) = o;
}

// ---------------------------------------------------------------------------
// Kernel 2: per-group mixed weights  W_g = sum_e c[g,e]*W_e + W_shared  (bf16)
// ---------------------------------------------------------------------------
__global__ __launch_bounds__(256) void mole_mix_w(const float* __restrict__ coeff,
                                                  const float* __restrict__ we,
                                                  const float* __restrict__ wsh,
                                                  unsigned short* __restrict__ mw_) {
  __bf16* mw = reinterpret_cast<__bf16*>(mw_);
  const int MAT = OUT_F * IN_F;            // 262144
  const int bpg = MAT / (256 * 8);         // 128
  int g   = blockIdx.x / bpg;
  int off = (blockIdx.x % bpg) * 2048 + threadIdx.x * 8;
  float c[NEXP];
#pragma unroll
  for (int e = 0; e < NEXP; ++e) c[e] = coeff[g * NEXP + e];
  f32x4 a0 = *(const f32x4*)(wsh + off);
  f32x4 a1 = *(const f32x4*)(wsh + off + 4);
#pragma unroll
  for (int e = 0; e < NEXP; ++e) {
    f32x4 w0 = *(const f32x4*)(we + (size_t)e * MAT + off);
    f32x4 w1 = *(const f32x4*)(we + (size_t)e * MAT + off + 4);
    a0 += c[e] * w0;
    a1 += c[e] * w1;
  }
  bf16x8 o;
#pragma unroll
  for (int i = 0; i < 4; ++i) { o[i] = (__bf16)a0[i]; o[i + 4] = (__bf16)a1[i]; }
  *(bf16x8*)(mw + (size_t)g * MAT + off) = o;
}

// ---------------------------------------------------------------------------
// Kernel 3: mixed bias  b_g = sum_e c[g,e]*b_e + b_shared  (f32, kept exact)
// ---------------------------------------------------------------------------
__global__ __launch_bounds__(256) void mole_mix_b(const float* __restrict__ coeff,
                                                  const float* __restrict__ be,
                                                  const float* __restrict__ bsh,
                                                  float* __restrict__ mb) {
  int idx = blockIdx.x * 256 + threadIdx.x;     // < NGRP*OUT_F = 16384
  int g = idx >> 9;
  int o = idx & (OUT_F - 1);
  float acc = bsh[o];
#pragma unroll
  for (int e = 0; e < NEXP; ++e) acc += coeff[g * NEXP + e] * be[e * OUT_F + o];
  mb[idx] = acc;
}

// ---------------------------------------------------------------------------
// Kernel 4: grouped GEMM  out[g] = x_g(bf16) @ W_g(bf16)^T + b_g   (f32 out)
// 128x128 tile / block, BK=64, double-buffered LDS fed by
// GLOBAL_LOAD_ASYNC_TO_LDS_B128 (ASYNCcnt pipeline), 8 waves (4x2),
// each wave: 2x4 tiles of v_wmma_f32_16x16x32_bf16.
// ---------------------------------------------------------------------------
__global__ __launch_bounds__(256) void mole_gemm(const unsigned short* __restrict__ xb_,
                                                 const unsigned short* __restrict__ mw_,
                                                 const float* __restrict__ mb,
                                                 float* __restrict__ out) {
  const __bf16* xb = reinterpret_cast<const __bf16*>(xb_);
  const __bf16* mw = reinterpret_cast<const __bf16*>(mw_);

  // one flat LDS arena: buf b -> [ A: BM*LDT | B: BN*LDT ]
  __shared__ __bf16 smem[2 * (BM + BN) * LDT];   // 81920 bytes

  const int tid = threadIdx.x;
  const int bid = blockIdx.x;
  const int nb  = bid & 3;             // OUT_F / BN = 4 column blocks
  const int tb  = bid >> 2;            // 512 token blocks
  const int g   = tb >> 4;             // TPG / BM = 16 token blocks per group
  const int m0  = tb * BM;
  const int n0  = nb * BN;

  const __bf16* Aglob = xb + (size_t)m0 * IN_F;
  const __bf16* Bglob = mw + (size_t)g * (OUT_F * IN_F) + (size_t)n0 * IN_F;

  // staging: 256 lanes x 4 passes x 16B each for A and B
  const int lrow = tid >> 3;           // 0..31
  const int lcol = (tid & 7) * 8;      // 0,8,..,56

  const unsigned smem_base = (unsigned)(unsigned long long)(&smem[0]);

  auto issue_async = [&](int buf, int k0) {
    unsigned abase = smem_base + (unsigned)(buf * (BM + BN) * LDT * 2);
    unsigned bbase = abase + (unsigned)(BM * LDT * 2);
#pragma unroll
    for (int it = 0; it < 4; ++it) {
      int r = lrow + it * 32;
      unsigned loff = (unsigned)((r * LDT + lcol) * 2);
      async_cp16(abase + loff, Aglob + (size_t)r * IN_F + k0 + lcol);
      async_cp16(bbase + loff, Bglob + (size_t)r * IN_F + k0 + lcol);
    }
  };

  // wave tiling: 4x2 wave grid, each wave 32(M) x 64(N) = 2x4 WMMA tiles
  const int w    = tid >> 5;
  const int wm   = w & 3;
  const int wn   = w >> 2;
  const int lane = tid & 31;
  const int half = lane >> 4;          // K-half select per A/B fragment layout
  const int lr   = lane & 15;          // row (A) / column (B) within tile

  f32x8 acc[2][4] = {};

  issue_async(0, 0);
  ASYNC_WAIT0();
  __syncthreads();

  int buf = 0;
  const int KT = IN_F / BK;            // 8
  for (int kt = 1; kt <= KT; ++kt) {
    if (kt < KT) issue_async(buf ^ 1, kt * BK);  // DMA next tile into idle buffer

    const __bf16* Ab = &smem[buf * (BM + BN) * LDT];
    const __bf16* Bb = Ab + BM * LDT;
#pragma unroll
    for (int kk = 0; kk < BK; kk += 32) {
      bf16x16 af[2];
      bf16x16 bfv[4];
#pragma unroll
      for (int mt = 0; mt < 2; ++mt)
        af[mt] = *(const bf16x16*)(Ab + (wm * 32 + mt * 16 + lr) * LDT + kk + half * 16);
#pragma unroll
      for (int nt = 0; nt < 4; ++nt)
        bfv[nt] = *(const bf16x16*)(Bb + (wn * 64 + nt * 16 + lr) * LDT + kk + half * 16);
#pragma unroll
      for (int mt = 0; mt < 2; ++mt)
#pragma unroll
        for (int nt = 0; nt < 4; ++nt)
          acc[mt][nt] = __builtin_amdgcn_wmma_f32_16x16x32_bf16(
              false, af[mt], false, bfv[nt], (short)0, acc[mt][nt], false, false);
    }

    if (kt < KT) ASYNC_WAIT0();        // this wave's DMAs into idle buffer done
    __syncthreads();                   // everyone's DMAs visible, all reads of buf done
    buf ^= 1;
  }

  // epilogue: C/D layout — VGPR j: lanes 0-15 -> M=j, lanes 16-31 -> M=8+j; N=lane&15
#pragma unroll
  for (int nt = 0; nt < 4; ++nt) {
    int col = n0 + wn * 64 + nt * 16 + lr;
    float bias = mb[g * OUT_F + col];
#pragma unroll
    for (int mt = 0; mt < 2; ++mt) {
      int rowb = m0 + wm * 32 + mt * 16 + half * 8;
#pragma unroll
      for (int j = 0; j < 8; ++j)
        out[(size_t)(rowb + j) * OUT_F + col] = acc[mt][nt][j] + bias;
    }
  }
}

// ---------------------------------------------------------------------------
extern "C" void kernel_launch(void* const* d_in, const int* in_sizes, int n_in,
                              void* d_out, int out_size, void* d_ws, size_t ws_size,
                              hipStream_t stream) {
  const float* x     = (const float*)d_in[0];
  const float* coeff = (const float*)d_in[1];
  const float* we    = (const float*)d_in[2];
  const float* be    = (const float*)d_in[3];
  const float* wsh   = (const float*)d_in[4];
  const float* bsh   = (const float*)d_in[5];
  // d_in[6] = sizes: static, all equal to TPG -> unused
  float* out = (float*)d_out;

  // workspace layout: [mixed W bf16: 16 MiB][mixed bias f32: 64 KiB][x bf16: 64 MiB]
  unsigned short* mw  = (unsigned short*)d_ws;
  float*          mb  = (float*)((char*)d_ws + (size_t)16 * 1024 * 1024);
  unsigned short* xbf = (unsigned short*)((char*)d_ws + (size_t)16 * 1024 * 1024 + 64 * 1024);

  mole_xcvt <<<NTOK * IN_F / (256 * 8), 256, 0, stream>>>(x, xbf);
  mole_mix_w<<<NGRP * (OUT_F * IN_F / (256 * 8)), 256, 0, stream>>>(coeff, we, wsh, mw);
  mole_mix_b<<<NGRP * OUT_F / 256, 256, 0, stream>>>(coeff, be, bsh, mb);
  mole_gemm <<<(NTOK / BM) * (OUT_F / BN), 256, 0, stream>>>(xbf, mw, mb, out);
}
